// ConsciousnessGuidedAttention_5540507812450
// MI455X (gfx1250) — matrix operand order repaired and checked by reading, not compiled, not run on hardware
//
#include <hip/hip_runtime.h>
#include <cstdint>
#include <cstddef>

#define BB 2
#define SS 2048
#define EE 1024
#define HH 16
#define LL 5
#define DHH 64
#define BH (BB*HH)

// GEMM tiling
#define GM 64
#define GN 256
#define GK 32
#define APAD 40   // padded LDS row stride (elements) -> conflict-free ds_read_b128
#define BPAD 40

typedef __bf16 bf16_t;
typedef bf16_t v16bf __attribute__((ext_vector_type(16)));
typedef bf16_t v8bf  __attribute__((ext_vector_type(8)));
typedef float  v8f   __attribute__((ext_vector_type(8)));

// ---------------------------------------------------------------------------
// WMMA helpers (CDNA5 wave32, v_wmma_f32_16x16x32_bf16)
// ---------------------------------------------------------------------------
__device__ __forceinline__ v8f wmma_bf16(v16bf a, v16bf b, v8f c) {
  return __builtin_amdgcn_wmma_f32_16x16x32_bf16(false, a, false, b,
                                                 (short)0, c, false, false);
}

// A fragment: 16x32 bf16 tile from row-major source (ld = lda), global or LDS.
// ISA layout: lane m=lane&15, kh=lane>>4; elems i<8 -> k=kh*8+i, i>=8 -> 16+kh*8+(i-8).
__device__ __forceinline__ v16bf load_a_frag(const bf16_t* A, int lda, int lane) {
  int m  = lane & 15;
  int kh = lane >> 4;
  const bf16_t* p = A + (size_t)m * lda + kh * 8;
  v8bf lo = *(const v8bf*)p;
  v8bf hi = *(const v8bf*)(p + 16);
  v16bf r;
#pragma unroll
  for (int i = 0; i < 8; ++i) { r[i] = lo[i]; r[i + 8] = hi[i]; }
  return r;
}

// B fragment: 32x16 (KxN) read from B^T stored row-major (N rows x K cols, ld=ldb).
// ISA layout: lane n=lane&15, kh=lane>>4; elems i<8 -> k=kh*8+i, i>=8 -> 16+kh*8+(i-8).
__device__ __forceinline__ v16bf load_b_frag(const bf16_t* BT, int ldb, int lane) {
  int n  = lane & 15;
  int kh = lane >> 4;
  const bf16_t* p = BT + (size_t)n * ldb + kh * 8;
  v8bf lo = *(const v8bf*)p;
  v8bf hi = *(const v8bf*)(p + 16);
  v16bf r;
#pragma unroll
  for (int i = 0; i < 8; ++i) { r[i] = lo[i]; r[i + 8] = hi[i]; }
  return r;
}

// ---------------------------------------------------------------------------
// Prep kernels
// ---------------------------------------------------------------------------
__global__ void k_prep_w(const float* __restrict__ Wc, const float* __restrict__ Wq,
                         const float* __restrict__ Wk, const float* __restrict__ Wv,
                         const float* __restrict__ Wo, const float* __restrict__ CL,
                         bf16_t* __restrict__ WcT, bf16_t* __restrict__ WqT,
                         bf16_t* __restrict__ WkT, bf16_t* __restrict__ WvT,
                         bf16_t* __restrict__ WoT)
{
  int idx = blockIdx.x * blockDim.x + threadIdx.x;
  if (idx >= EE * EE) return;
  int e = idx / EE, o = idx - e * EE;
  size_t src = (size_t)e * EE + o;   // W[e][o]
  size_t dst = (size_t)o * EE + e;   // WT[o][e]
  WqT[dst] = (bf16_t)Wq[src];
  WkT[dst] = (bf16_t)Wk[src];
  WvT[dst] = (bf16_t)Wv[src];
  WoT[dst] = (bf16_t)Wo[src];
  float s0 = 0.f, s1 = 0.f;
#pragma unroll
  for (int l = 0; l < LL; ++l) {
    float w = Wc[(size_t)l * EE * EE + src];
    s0 += (CL[0 * HH + l] * (1.0f / LL)) * w;   // l % H == l since L < H
    s1 += (CL[1 * HH + l] * (1.0f / LL)) * w;
  }
  WcT[dst] = (bf16_t)s0;
  WcT[(size_t)EE * EE + dst] = (bf16_t)s1;
}

__global__ void k_bceff(const float* __restrict__ bc, const float* __restrict__ CL,
                        float* __restrict__ bce)
{
  int idx = blockIdx.x * blockDim.x + threadIdx.x;
  if (idx >= BB * EE) return;
  int bz = idx / EE, e = idx - bz * EE;
  float s = 0.f;
#pragma unroll
  for (int l = 0; l < LL; ++l)
    s += CL[bz * HH + l] * (1.0f / LL) * bc[l * EE + e];
  bce[idx] = s;
}

__global__ void k_x2bf(const float* __restrict__ x, bf16_t* __restrict__ x16)
{
  int idx = blockIdx.x * blockDim.x + threadIdx.x;
  if (idx < BB * SS * EE) x16[idx] = (bf16_t)x[idx];
}

// ---------------------------------------------------------------------------
// LDS-staged double-buffered bf16 WMMA GEMM:
//   C[M,N] = A[M,K] @ B[K,N] + bias, with B supplied as B^T (N rows x K cols).
// Block tile 64(M) x 256(N), k-step 32. 8 waves in 2x4; each wave 32x64
// (2x4 wmma tiles, 8 accumulators). Next k-step is fetched to registers
// before the WMMAs so vmem latency overlaps the matrix pipe.
// ---------------------------------------------------------------------------
__global__ void __launch_bounds__(256)
k_gemm(const bf16_t* __restrict__ A, size_t strideA, int lda,
       const bf16_t* __restrict__ BT, size_t strideB, int ldb,
       const float* __restrict__ bias, size_t strideBias,
       float* __restrict__ C32, bf16_t* __restrict__ C16,
       size_t strideC, int ldc, int K)
{
  __shared__ __align__(16) bf16_t As[2][GM * APAD];   // 10.0 KB
  __shared__ __align__(16) bf16_t Bs[2][GN * BPAD];   // 40.0 KB
  int z = blockIdx.z;
  A  += (size_t)z * strideA;
  BT += (size_t)z * strideB;
  const float* bp = bias ? bias + (size_t)z * strideBias : nullptr;

  int tid = threadIdx.x, lane = tid & 31, wid = tid >> 5;
  int m0 = blockIdx.y * GM;
  int n0 = blockIdx.x * GN;
  int wm = (wid & 1) * 32;     // wave tile offset in block (2 m-waves)
  int wn = (wid >> 1) * 64;    // 4 n-waves

  // staging: A tile 64x32 = 256 16B-chunks (1/thread); B tile 256x32 = 1024 (4/thread)
  int arow = tid >> 2, achk = (tid & 3) * 8;
  const bf16_t* agp = A + (size_t)(m0 + arow) * lda + achk;
  int brow[4], bchk[4];
  const bf16_t* bgp[4];
#pragma unroll
  for (int i = 0; i < 4; ++i) {
    int cidx = tid + 256 * i;
    brow[i] = cidx >> 2;
    bchk[i] = (cidx & 3) * 8;
    bgp[i] = BT + (size_t)(n0 + brow[i]) * ldb + bchk[i];
  }

  v8bf areg = *(const v8bf*)agp;
  v8bf breg0 = *(const v8bf*)bgp[0];
  v8bf breg1 = *(const v8bf*)bgp[1];
  v8bf breg2 = *(const v8bf*)bgp[2];
  v8bf breg3 = *(const v8bf*)bgp[3];
  *(v8bf*)&As[0][arow * APAD + achk] = areg;
  *(v8bf*)&Bs[0][brow[0] * BPAD + bchk[0]] = breg0;
  *(v8bf*)&Bs[0][brow[1] * BPAD + bchk[1]] = breg1;
  *(v8bf*)&Bs[0][brow[2] * BPAD + bchk[2]] = breg2;
  *(v8bf*)&Bs[0][brow[3] * BPAD + bchk[3]] = breg3;
  __syncthreads();

  v8f acc[2][4] = {};
  int nk = K / GK;
  for (int kb = 0; kb < nk; ++kb) {
    int cur = kb & 1, nxt = cur ^ 1;
    if (kb + 1 < nk) {
      int ko = (kb + 1) * GK;
      areg  = *(const v8bf*)(agp + ko);
      breg0 = *(const v8bf*)(bgp[0] + ko);
      breg1 = *(const v8bf*)(bgp[1] + ko);
      breg2 = *(const v8bf*)(bgp[2] + ko);
      breg3 = *(const v8bf*)(bgp[3] + ko);
    }
    v16bf af[2], bfrag[4];
#pragma unroll
    for (int mi = 0; mi < 2; ++mi)
      af[mi] = load_a_frag(&As[cur][(wm + mi * 16) * APAD], APAD, lane);
#pragma unroll
    for (int ni = 0; ni < 4; ++ni)
      bfrag[ni] = load_b_frag(&Bs[cur][(wn + ni * 16) * BPAD], BPAD, lane);
#pragma unroll
    for (int mi = 0; mi < 2; ++mi)
#pragma unroll
      for (int ni = 0; ni < 4; ++ni)
        acc[mi][ni] = wmma_bf16(af[mi], bfrag[ni], acc[mi][ni]);
    __syncthreads();
    if (kb + 1 < nk) {
      *(v8bf*)&As[nxt][arow * APAD + achk] = areg;
      *(v8bf*)&Bs[nxt][brow[0] * BPAD + bchk[0]] = breg0;
      *(v8bf*)&Bs[nxt][brow[1] * BPAD + bchk[1]] = breg1;
      *(v8bf*)&Bs[nxt][brow[2] * BPAD + bchk[2]] = breg2;
      *(v8bf*)&Bs[nxt][brow[3] * BPAD + bchk[3]] = breg3;
    }
    __syncthreads();
  }

  int n = lane & 15, mh = lane >> 4;
  float*  c32 = C32 ? C32 + (size_t)z * strideC : nullptr;
  bf16_t* c16 = C16 ? C16 + (size_t)z * strideC : nullptr;
#pragma unroll
  for (int mi = 0; mi < 2; ++mi)
#pragma unroll
    for (int ni = 0; ni < 4; ++ni)
#pragma unroll
      for (int g = 0; g < 8; ++g) {
        int row = m0 + wm + mi * 16 + mh * 8 + g;
        int col = n0 + wn + ni * 16 + n;
        float v = acc[mi][ni][g];
        if (bp) v += bp[col];
        size_t o = (size_t)row * ldc + col;
        if (c32) c32[o] = v;
        if (c16) c16[o] = (bf16_t)v;
      }
}

// ---------------------------------------------------------------------------
// Column means over the S axis:  out[b,e] = mean_s in[b,s,e]
// ---------------------------------------------------------------------------
__global__ void k_colmean_f32(const float* __restrict__ in, float* __restrict__ out,
                              int rows, int cols, int nb)
{
  int idx = blockIdx.x * blockDim.x + threadIdx.x;
  if (idx >= nb * cols) return;
  int bz = idx / cols, e = idx - bz * cols;
  const float* p = in + (size_t)bz * rows * cols + e;
  float s = 0.f;
  for (int i = 0; i < rows; ++i) s += p[(size_t)i * cols];
  out[idx] = s / (float)rows;
}

__global__ void k_colmean_bf16(const bf16_t* __restrict__ in, float* __restrict__ out,
                               int rows, int cols, int nb)
{
  int idx = blockIdx.x * blockDim.x + threadIdx.x;
  if (idx >= nb * cols) return;
  int bz = idx / cols, e = idx - bz * cols;
  const bf16_t* p = in + (size_t)bz * rows * cols + e;
  float s = 0.f;
  for (int i = 0; i < rows; ++i) s += (float)p[(size_t)i * cols];
  out[idx] = s / (float)rows;
}

// ---------------------------------------------------------------------------
// Tiled bf16 transpose per batch: out[e][s] = in[s][e]
// ---------------------------------------------------------------------------
__global__ void __launch_bounds__(256)
k_transp(const bf16_t* __restrict__ in, bf16_t* __restrict__ out, int rows, int cols)
{
  __shared__ bf16_t tile[32][33];
  int z = blockIdx.z;
  const bf16_t* ip = in + (size_t)z * rows * cols;
  bf16_t* op = out + (size_t)z * rows * cols;
  int c0 = blockIdx.x * 32, r0 = blockIdx.y * 32;
  int tx = threadIdx.x & 31, ty = threadIdx.x >> 5;
  for (int i = ty; i < 32; i += 8) tile[i][tx] = ip[(size_t)(r0 + i) * cols + c0 + tx];
  __syncthreads();
  for (int i = ty; i < 32; i += 8) op[(size_t)(c0 + i) * rows + r0 + tx] = tile[tx][i];
}

// ---------------------------------------------------------------------------
// Per-(b,h) scalars: scoreScale=(1+cw)/8, scoreBias=0.1*phase*(1+cw), factor
// ---------------------------------------------------------------------------
__global__ void __launch_bounds__(256)
k_small(const float* __restrict__ pooled, const float* __restrict__ Qm,
        const float* __restrict__ Km, const float* __restrict__ Wf,
        const float* __restrict__ bfp, const float* __restrict__ Wc1,
        const float* __restrict__ bc1, const float* __restrict__ Wc2,
        const float* __restrict__ bc2, const float* __restrict__ gate,
        const float* __restrict__ phi, const float* __restrict__ CL,
        float* __restrict__ sstat)
{
  __shared__ float red[256];
  __shared__ float cwsh;
  __shared__ float freq[LL];
  int bh = blockIdx.x;
  int b = bh >> 4, h = bh & 15;
  int tid = threadIdx.x;

  // cw = sigmoid( gelu(ci @ Wc1 + bc1) @ Wc2 + bc2 ),  ci = [Qmean_h | Kmean_h]
  const float* qv = Qm + (size_t)b * EE + h * DHH;
  const float* kv = Km + (size_t)b * EE + h * DHH;
  float part = 0.f;
  for (int j = tid; j < EE; j += 256) {
    float hs = bc1[j];
    for (int i = 0; i < DHH; ++i)
      hs += qv[i] * Wc1[(size_t)i * EE + j] + kv[i] * Wc1[(size_t)(DHH + i) * EE + j];
    float g = 0.5f * hs * (1.0f + erff(hs * 0.70710678f));
    part += g * Wc2[j];
  }
  red[tid] = part; __syncthreads();
  for (int st = 128; st > 0; st >>= 1) { if (tid < st) red[tid] += red[tid + st]; __syncthreads(); }
  if (tid == 0) cwsh = 1.0f / (1.0f + expf(-(red[0] + bc2[0])));
  __syncthreads();

  // freq[l] = pooled[b] . Wf[l,:,h] + bf[l,h]
  for (int l = 0; l < LL; ++l) {
    float p = 0.f;
    for (int e = tid; e < EE; e += 256)
      p += pooled[(size_t)b * EE + e] * Wf[((size_t)l * EE + e) * HH + h];
    red[tid] = p; __syncthreads();
    for (int st = 128; st > 0; st >>= 1) { if (tid < st) red[tid] += red[tid + st]; __syncthreads(); }
    if (tid == 0) freq[l] = red[0] + bfp[l * HH + h];
    __syncthreads();
  }

  if (tid == 0) {
    float onep = 1.0f + cwsh;
    float ph = 0.f;
    for (int l = 0; l < LL; ++l)
      ph += sinf(phi[h] * freq[l] * CL[b * HH + l]);
    ph *= (1.0f / LL);
    float fac = 1.0f;
    for (int l = 0; l < LL; ++l) {
      float den = 0.f;
      for (int hh = 0; hh < HH; ++hh) den += expf(gate[l * HH + hh]);
      float gw = expf(gate[l * HH + h]) / den;
      fac *= 1.0f + 0.1f * CL[b * HH + l] * gw;
    }
    sstat[bh]          = onep * 0.125f;      // (1+cw)/sqrt(64)
    sstat[BH + bh]     = 0.1f * ph * onep;   // additive term, pre-scaled by (1+cw)
    sstat[2 * BH + bh] = fac;
  }
}

// ---------------------------------------------------------------------------
// Attention: one block per (b, h, 16-row tile). Score strip in LDS
// (CDNA5: 320KB/WGP, ~140KB used). Double softmax in-place; PV via WMMA with
// a deterministic 2-way K-split reduction.
// ---------------------------------------------------------------------------
__global__ void __launch_bounds__(256)
k_attn(const bf16_t* __restrict__ Qf, const bf16_t* __restrict__ Kf,
       const bf16_t* __restrict__ VT, const float* __restrict__ sstat,
       bf16_t* __restrict__ att)
{
  __shared__ float sc[16][SS + 1];      // ~131KB score strip
  __shared__ float part[2][16][DHH];    // PV partials (K-split)
  __shared__ float red[16][16];
  __shared__ float rowm[16], rows1[16], roww[16];

  const int nb = SS / 16;
  int bh = blockIdx.x / nb;
  int rb = blockIdx.x % nb;
  int b = bh >> 4, h = bh & 15;
  int r0 = rb * 16;
  float sScale = sstat[bh];
  float sBias  = sstat[BH + bh];
  float fac    = sstat[2 * BH + bh];
  int tid = threadIdx.x, lane = tid & 31, wid = tid >> 5;
  int n = lane & 15, mh = lane >> 4;

  // ---- scores = (Q K^T) * sScale + sBias ----
  const bf16_t* Qbase = Qf + ((size_t)b * SS + r0) * EE + h * DHH;
  const bf16_t* Kbase = Kf + (size_t)b * SS * EE + h * DHH;
  v16bf aq0 = load_a_frag(Qbase, EE, lane);
  v16bf aq1 = load_a_frag(Qbase + 32, EE, lane);
  for (int cb = wid; cb < nb; cb += 8) {
    int c0 = cb * 16;
    if (cb + 8 < nb)   // prefetch next key block for this wave
      __builtin_prefetch(Kbase + (size_t)(c0 + 128) * EE, 0, 1);
    v16bf kb0 = load_b_frag(Kbase + (size_t)c0 * EE, EE, lane);
    v16bf kb1 = load_b_frag(Kbase + (size_t)c0 * EE + 32, EE, lane);
    v8f acc = {};
    acc = wmma_bf16(aq0, kb0, acc);
    acc = wmma_bf16(aq1, kb1, acc);
#pragma unroll
    for (int g = 0; g < 8; ++g)
      sc[mh * 8 + g][c0 + n] = acc[g] * sScale + sBias;
  }
  __syncthreads();

  // ---- double softmax over each of the 16 rows ----
  int r = tid & 15, c = tid >> 4;
  float mx = -3.0e38f;
  for (int j = c; j < SS; j += 16) mx = fmaxf(mx, sc[r][j]);
  red[r][c] = mx; __syncthreads();
  if (c == 0) { float m2 = red[r][0]; for (int k = 1; k < 16; ++k) m2 = fmaxf(m2, red[r][k]); rowm[r] = m2; }
  __syncthreads();
  float rm = rowm[r];
  float s1 = 0.f;
  for (int j = c; j < SS; j += 16) s1 += expf(sc[r][j] - rm);
  red[r][c] = s1; __syncthreads();
  if (c == 0) { float t = 0.f; for (int k = 0; k < 16; ++k) t += red[r][k]; rows1[r] = t; }
  __syncthreads();
  // aw_i = e_i / s1;  w_i = fac*aw_i;  max_i w_i = fac/s1 (since max e_i = 1)
  float fs = fac / rows1[r];
  float sw = 0.f;
  for (int j = c; j < SS; j += 16) {
    float e = expf(sc[r][j] - rm);
    float t = fs * (e - 1.0f);     // w_i - wmax
    sc[r][j] = t;
    sw += expf(t);
  }
  red[r][c] = sw; __syncthreads();
  if (c == 0) { float t = 0.f; for (int k = 0; k < 16; ++k) t += red[r][k]; roww[r] = t; }
  __syncthreads();
  float invw = 1.0f / roww[r];
  for (int j = c; j < SS; j += 16) sc[r][j] = expf(sc[r][j]) * invw;
  __syncthreads();

  // ---- attended = P @ V  (P: 16 x SS in LDS, V via VT: rows = head dims) ----
  int wn = wid & 3;     // d-tile (4 x 16 = 64)
  int wk = wid >> 2;    // K-split half
  const bf16_t* Vb = VT + ((size_t)b * EE + h * DHH + wn * 16) * SS;
  v8f oacc = {};
  int m = lane & 15, kh = lane >> 4;
  for (int kc = wk; kc < SS / 32; kc += 2) {
    int t0 = kc * 32;
    v16bf pa;
#pragma unroll
    for (int i = 0; i < 8; ++i) {
      pa[i]     = (bf16_t)sc[m][t0 + kh * 8 + i];
      pa[i + 8] = (bf16_t)sc[m][t0 + 16 + kh * 8 + i];
    }
    v16bf vb = load_b_frag(Vb + t0, SS, lane);
    oacc = wmma_bf16(pa, vb, oacc);
  }
#pragma unroll
  for (int g = 0; g < 8; ++g)
    part[wk][mh * 8 + g][wn * 16 + n] = oacc[g];
  __syncthreads();
  for (int idx = tid; idx < 16 * DHH; idx += 256) {
    int mm = idx >> 6, dd = idx & 63;
    float v = part[0][mm][dd] + part[1][mm][dd];
    att[((size_t)b * SS + r0 + mm) * EE + h * DHH + dd] = (bf16_t)v;
  }
}

// ---------------------------------------------------------------------------
// Residual + LayerNorm: out = LN(x + y0) * g + b
// ---------------------------------------------------------------------------
__global__ void __launch_bounds__(256)
k_ln(const float* __restrict__ x, const float* __restrict__ y0,
     const float* __restrict__ g, const float* __restrict__ bl,
     float* __restrict__ out)
{
  __shared__ float red[256];
  __shared__ float mu_s, rstd_s;
  int row = blockIdx.x, tid = threadIdx.x;
  const float* xr = x  + (size_t)row * EE;
  const float* yr = y0 + (size_t)row * EE;
  float s = 0.f;
  for (int e = tid; e < EE; e += 256) s += xr[e] + yr[e];
  red[tid] = s; __syncthreads();
  for (int st = 128; st > 0; st >>= 1) { if (tid < st) red[tid] += red[tid + st]; __syncthreads(); }
  if (tid == 0) mu_s = red[0] * (1.0f / EE);
  __syncthreads();
  float mu = mu_s;
  float v = 0.f;
  for (int e = tid; e < EE; e += 256) { float d = xr[e] + yr[e] - mu; v += d * d; }
  red[tid] = v; __syncthreads();
  for (int st = 128; st > 0; st >>= 1) { if (tid < st) red[tid] += red[tid + st]; __syncthreads(); }
  if (tid == 0) rstd_s = rsqrtf(red[0] * (1.0f / EE) + 1e-5f);
  __syncthreads();
  float rs = rstd_s;
  float* o = out + (size_t)row * EE;
  for (int e = tid; e < EE; e += 256)
    o[e] = (xr[e] + yr[e] - mu) * rs * g[e] + bl[e];
}

// ---------------------------------------------------------------------------
// Launcher
// ---------------------------------------------------------------------------
extern "C" void kernel_launch(void* const* d_in, const int* in_sizes, int n_in,
                              void* d_out, int out_size, void* d_ws, size_t ws_size,
                              hipStream_t stream)
{
  (void)in_sizes; (void)n_in; (void)out_size; (void)ws_size;
  const float* x    = (const float*)d_in[0];
  const float* CL   = (const float*)d_in[1];
  const float* Wc   = (const float*)d_in[2];
  const float* bc   = (const float*)d_in[3];
  const float* Wf   = (const float*)d_in[4];
  const float* bfp  = (const float*)d_in[5];
  const float* Wq   = (const float*)d_in[6];
  const float* bq   = (const float*)d_in[7];
  const float* Wk   = (const float*)d_in[8];
  const float* bk   = (const float*)d_in[9];
  const float* Wv   = (const float*)d_in[10];
  const float* bv   = (const float*)d_in[11];
  const float* Wo   = (const float*)d_in[12];
  const float* bo   = (const float*)d_in[13];
  const float* Wc1  = (const float*)d_in[14];
  const float* bc1  = (const float*)d_in[15];
  const float* Wc2  = (const float*)d_in[16];
  const float* bc2  = (const float*)d_in[17];
  const float* gate = (const float*)d_in[18];
  const float* phi  = (const float*)d_in[19];
  const float* lng  = (const float*)d_in[20];
  const float* lnb  = (const float*)d_in[21];
  float* out = (float*)d_out;

  char* ws = (char*)d_ws;
  size_t off = 0;
  auto alloc = [&](size_t bytes) -> char* {
    char* p = ws + off;
    off = (off + bytes + 255) & ~(size_t)255;
    return p;
  };
  const size_t NTOK = (size_t)BB * SS * EE;
  bf16_t* WcT   = (bf16_t*)alloc((size_t)BB * EE * EE * 2);
  bf16_t* WqT   = (bf16_t*)alloc((size_t)EE * EE * 2);
  bf16_t* WkT   = (bf16_t*)alloc((size_t)EE * EE * 2);
  bf16_t* WvT   = (bf16_t*)alloc((size_t)EE * EE * 2);
  bf16_t* WoT   = (bf16_t*)alloc((size_t)EE * EE * 2);
  float*  bce   = (float*)alloc((size_t)BB * EE * 4);
  bf16_t* x16   = (bf16_t*)alloc(NTOK * 2);
  float*  ce32  = (float*)alloc(NTOK * 4);
  bf16_t* ce16  = (bf16_t*)alloc(NTOK * 2);
  bf16_t* Qf    = (bf16_t*)alloc(NTOK * 2);
  bf16_t* Kf    = (bf16_t*)alloc(NTOK * 2);
  bf16_t* Vf    = (bf16_t*)alloc(NTOK * 2);
  bf16_t* VTb   = (bf16_t*)alloc(NTOK * 2);
  bf16_t* att16 = (bf16_t*)alloc(NTOK * 2);
  float*  y0    = (float*)alloc(NTOK * 4);
  float*  pooled= (float*)alloc((size_t)BB * EE * 4);
  float*  Qm    = (float*)alloc((size_t)BB * EE * 4);
  float*  Km    = (float*)alloc((size_t)BB * EE * 4);
  float*  sstat = (float*)alloc((size_t)3 * BH * 4);

  // 1) effective / transposed weights, bias, bf16 x
  k_prep_w<<<(EE * EE + 255) / 256, 256, 0, stream>>>(Wc, Wq, Wk, Wv, Wo, CL,
                                                      WcT, WqT, WkT, WvT, WoT);
  k_bceff<<<(BB * EE + 255) / 256, 256, 0, stream>>>(bc, CL, bce);
  k_x2bf<<<(int)((NTOK + 255) / 256), 256, 0, stream>>>(x, x16);

  // 2) ce = x @ Wc_eff[b] + bc_eff[b]   (per-batch GEMM, fp32 + bf16 outputs)
  k_gemm<<<dim3(EE / GN, SS / GM, BB), 256, 0, stream>>>(
      x16, (size_t)SS * EE, EE, WcT, (size_t)EE * EE, EE,
      bce, (size_t)EE, ce32, ce16, (size_t)SS * EE, EE, EE);

  // 3) pooled = mean_s ce
  k_colmean_f32<<<(BB * EE + 255) / 256, 256, 0, stream>>>(ce32, pooled, SS, EE, BB);

  // 4) Q/K/V projections (bf16 outputs)
  k_gemm<<<dim3(EE / GN, (BB * SS) / GM, 1), 256, 0, stream>>>(
      ce16, 0, EE, WqT, 0, EE, bq, 0, (float*)nullptr, Qf, 0, EE, EE);
  k_gemm<<<dim3(EE / GN, (BB * SS) / GM, 1), 256, 0, stream>>>(
      ce16, 0, EE, WkT, 0, EE, bk, 0, (float*)nullptr, Kf, 0, EE, EE);
  k_gemm<<<dim3(EE / GN, (BB * SS) / GM, 1), 256, 0, stream>>>(
      ce16, 0, EE, WvT, 0, EE, bv, 0, (float*)nullptr, Vf, 0, EE, EE);

  // 5) Q/K means, V transpose for PV B-fragments
  k_colmean_bf16<<<(BB * EE + 255) / 256, 256, 0, stream>>>(Qf, Qm, SS, EE, BB);
  k_colmean_bf16<<<(BB * EE + 255) / 256, 256, 0, stream>>>(Kf, Km, SS, EE, BB);
  k_transp<<<dim3(EE / 32, SS / 32, BB), 256, 0, stream>>>(Vf, VTb, SS, EE);

  // 6) per-(b,h) scalars
  k_small<<<BH, 256, 0, stream>>>(pooled, Qm, Km, Wf, bfp, Wc1, bc1, Wc2, bc2,
                                  gate, phi, CL, sstat);

  // 7) attention (double softmax) -> attended (bf16)
  k_attn<<<BH * (SS / 16), 256, 0, stream>>>(Qf, Kf, VTb, sstat, att16);

  // 8) output projection
  k_gemm<<<dim3(EE / GN, (BB * SS) / GM, 1), 256, 0, stream>>>(
      att16, 0, EE, WoT, 0, EE, bo, 0, y0, (bf16_t*)nullptr, 0, EE, EE);

  // 9) residual + layernorm
  k_ln<<<BB * SS, 256, 0, stream>>>(x, y0, lng, lnb, out);
}
